// Model_89326729822211
// MI455X (gfx1250) — compile-verified, matrix operand order checked by reading, not run on hardware
//
#include <hip/hip_runtime.h>
#include <hip/hip_bf16.h>

typedef __attribute__((ext_vector_type(16))) _Float16 v16h;
typedef __attribute__((ext_vector_type(8)))  float    v8f;

#define BATCH 2
#define LSEQ  256
#define TOK   512   // BATCH*LSEQ
#define DMODEL 256
#define DINNER 512
#define NSTATE 16
#define XPROJ_N 48  // R + 2N

// ---------------------------------------------------------------------------
// Generic f16 WMMA GEMM:  C[TOK x N] = A[TOK x K] * Bt[N x K]^T (+bias)(+Cadd)
// one 16x16 tile per wave, 4 waves per block; grid = (N/16, TOK/64)
// ---------------------------------------------------------------------------
__global__ void gemm_wmma(const _Float16* __restrict__ A,
                          const _Float16* __restrict__ Bt,
                          const float* __restrict__ bias,
                          const float* __restrict__ Cadd,
                          float* __restrict__ C, int N, int K) {
  const int lane = threadIdx.x & 31;
  const int wave = threadIdx.x >> 5;
  const int tm = (blockIdx.y * 4 + wave) << 4;
  const int tn = blockIdx.x << 4;
  const int am = tm + (lane & 15);
  const int bn = tn + (lane & 15);
  const int kb = (lane >> 4) << 4;     // lanes 0-15: K 0..15, lanes 16-31: K 16..31
  const int cm = tm + ((lane >> 4) << 3);
  const int cn = tn + (lane & 15);

  v8f c = {};
  if (Cadd) {
#pragma unroll
    for (int r = 0; r < 8; ++r) c[r] = Cadd[(size_t)(cm + r) * N + cn];
  }
  const _Float16* ap = A  + (size_t)am * K + kb;
  const _Float16* bp = Bt + (size_t)bn * K + kb;
  for (int k0 = 0; k0 < K; k0 += 32) {
    v16h a = *(const v16h*)(ap + k0);
    v16h b = *(const v16h*)(bp + k0);
    c = __builtin_amdgcn_wmma_f32_16x16x32_f16(false, a, false, b, (short)0, c,
                                               false, false);
  }
  const float bv = bias ? bias[cn] : 0.0f;
#pragma unroll
  for (int r = 0; r < 8; ++r) C[(size_t)(cm + r) * N + cn] = c[r] + bv;
}

// f32 [K][N] -> f16 transposed [N][K]
__global__ void convT_kernel(const float* __restrict__ W, _Float16* __restrict__ Wt,
                             int K, int N) {
  int i = blockIdx.x * blockDim.x + threadIdx.x;
  if (i >= K * N) return;
  int k = i / N, n = i - k * N;
  Wt[(size_t)n * K + k] = (_Float16)W[i];
}

__global__ void cast16_kernel(const float* __restrict__ X, _Float16* __restrict__ Y, int n) {
  int i = blockIdx.x * blockDim.x + threadIdx.x;
  if (i < n) Y[i] = (_Float16)X[i];
}

__global__ void add_kernel(float* __restrict__ X, const float* __restrict__ Y, int n) {
  int i = blockIdx.x * blockDim.x + threadIdx.x;
  if (i < n) X[i] += Y[i];
}

// Row LayerNorm over D=256 (one block of 256 threads per row), optional silu,
// optional f32 and/or f16 outputs.
__global__ void ln_kernel(const float* __restrict__ X, const float* __restrict__ g,
                          const float* __restrict__ b, float* __restrict__ o32,
                          _Float16* __restrict__ o16, int do_silu) {
  __shared__ float red[DMODEL];
  const int row = blockIdx.x, tid = threadIdx.x;
  float x = X[(size_t)row * DMODEL + tid];
  red[tid] = x; __syncthreads();
  for (int s = 128; s > 0; s >>= 1) { if (tid < s) red[tid] += red[tid + s]; __syncthreads(); }
  const float mean = red[0] * (1.0f / DMODEL);
  __syncthreads();
  const float dv = x - mean;
  red[tid] = dv * dv; __syncthreads();
  for (int s = 128; s > 0; s >>= 1) { if (tid < s) red[tid] += red[tid + s]; __syncthreads(); }
  const float var = red[0] * (1.0f / DMODEL);
  float y = dv * rsqrtf(var + 1e-5f) * g[tid] + b[tid];
  if (do_silu) y = y * (1.0f / (1.0f + __expf(-y)));
  if (o32) o32[(size_t)row * DMODEL + tid] = y;
  if (o16) o16[(size_t)row * DMODEL + tid] = (_Float16)y;
}

// causal depthwise conv (DC=4) over l within each batch, + bias, + silu
// xz layout: [tok][1024], channel d of xm at column d; writes xm f32 and f16
__global__ void dwconv_kernel(const float* __restrict__ xz, const float* __restrict__ w,
                              const float* __restrict__ bias, float* __restrict__ xm32,
                              _Float16* __restrict__ xm16) {
  int idx = blockIdx.x * blockDim.x + threadIdx.x;  // TOK*DINNER
  if (idx >= TOK * DINNER) return;
  const int d = idx & (DINNER - 1);
  const int tok = idx >> 9;
  const int l = tok & (LSEQ - 1);
  float acc = bias[d];
#pragma unroll
  for (int j = 0; j < 4; ++j) {
    int ls = l - 3 + j;
    if (ls >= 0) acc += w[d * 4 + j] * xz[(size_t)(tok - 3 + j) * (2 * DINNER) + d];
  }
  float s = acc * (1.0f / (1.0f + __expf(-acc)));
  xm32[idx] = s;
  xm16[idx] = (_Float16)s;
}

// delta = softplus(x_dbl[:, :16] @ dt_w + dt_b), dt_w is [16][512]
__global__ void delta_kernel(const float* __restrict__ xdbl, const float* __restrict__ dt_w,
                             const float* __restrict__ dt_b, float* __restrict__ delta) {
  int idx = blockIdx.x * blockDim.x + threadIdx.x;  // TOK*DINNER
  if (idx >= TOK * DINNER) return;
  const int d = idx & (DINNER - 1);
  const int tok = idx >> 9;
  float acc = dt_b[d];
  const float* xr = xdbl + (size_t)tok * XPROJ_N;
#pragma unroll
  for (int r = 0; r < 16; ++r) acc += xr[r] * dt_w[r * DINNER + d];
  delta[idx] = (acc > 20.0f) ? acc : log1pf(__expf(acc));
}

// Selective scan: one lane per (b,d) chain, 16 SSM states in registers.
// Fuses exp(dA), C-dot, u*Dp, gating by silu(z), and cast to f16.
__global__ void scan_kernel(const float* __restrict__ xm32, const float* __restrict__ delta,
                            const float* __restrict__ xdbl, const float* __restrict__ A_log,
                            const float* __restrict__ Dp, const float* __restrict__ xz,
                            _Float16* __restrict__ yg16) {
  int idx = blockIdx.x * blockDim.x + threadIdx.x;  // BATCH*DINNER = 1024
  if (idx >= BATCH * DINNER) return;
  const int d = idx & (DINNER - 1);
  const int b = idx >> 9;
  float A[NSTATE], s[NSTATE];
#pragma unroll
  for (int n = 0; n < NSTATE; ++n) { A[n] = -__expf(A_log[d * NSTATE + n]); s[n] = 0.0f; }
  const float Dv = Dp[d];
  for (int l = 0; l < LSEQ; ++l) {
    const int tok = b * LSEQ + l;
    const float* xr = xdbl + (size_t)tok * XPROJ_N;
    __builtin_prefetch(xr + XPROJ_N, 0, 1);
    const float u = xm32[(size_t)tok * DINNER + d];
    const float dl = delta[(size_t)tok * DINNER + d];
    const float du = dl * u;
    float acc = 0.0f;
#pragma unroll
    for (int n = 0; n < NSTATE; ++n) {
      float a = __expf(dl * A[n]);
      s[n] = a * s[n] + du * xr[16 + n];
      acc += s[n] * xr[32 + n];
    }
    const float y = acc + u * Dv;
    const float z = xz[(size_t)tok * (2 * DINNER) + DINNER + d];
    const float gz = z * (1.0f / (1.0f + __expf(-z)));
    yg16[(size_t)tok * DINNER + d] = (_Float16)(y * gz);
  }
}

// pe[tok][192] gather from x (t<2) / fx (t>=2), cast to f16
__global__ void patch_kernel(const float* __restrict__ x, const float* __restrict__ fx,
                             _Float16* __restrict__ pe) {
  int idx = blockIdx.x * blockDim.x + threadIdx.x;  // TOK*192
  if (idx >= TOK * 192) return;
  const int kk = idx % 192;
  const int tok = idx / 192;
  const int b = tok >> 8, l = tok & 255;
  const int hh = l >> 4, ww = l & 15;
  const int pt = kk >> 4, rem = kk & 15, ph = rem >> 2, pw = rem & 3;
  const int h = hh * 4 + ph, w = ww * 4 + pw;
  const int pix = b * 4096 + h * 64 + w;
  const float v = (pt < 2) ? x[pix * 2 + pt] : fx[pix * 10 + (pt - 2)];
  pe[idx] = (_Float16)v;
}

// Scatter head GEMM output [512][16] into (B,64,64) + head_b
__global__ void head_kernel(const float* __restrict__ Yh, const float* __restrict__ hb,
                            float* __restrict__ out) {
  int idx = blockIdx.x * blockDim.x + threadIdx.x;  // 8192
  if (idx >= BATCH * 64 * 64) return;
  const int b = idx >> 12, rem = idx & 4095;
  const int h = rem >> 6, w = rem & 63;
  const int hh = h >> 2, ph = h & 3, ww = w >> 2, pw = w & 3;
  out[idx] = Yh[(size_t)((b << 8) + hh * 16 + ww) * 16 + ph * 4 + pw] + hb[0];
}

// ---------------------------------------------------------------------------
struct BlockP {
  const float *ln_g, *ln_b, *in_w, *conv_w, *conv_b, *xproj_w, *dt_w, *dt_b,
              *A_log, *Dp, *out_w;
};
struct LevelP {
  BlockP blk[2];
  const float *lin_w, *lin_b, *ln_g, *ln_b;
};

extern "C" void kernel_launch(void* const* d_in, const int* in_sizes, int n_in,
                              void* d_out, int out_size, void* d_ws, size_t ws_size,
                              hipStream_t stream) {
  (void)n_in; (void)out_size; (void)ws_size;
  const float* x_in  = (const float*)d_in[0];
  const float* fx_in = (const float*)d_in[1];

  auto F = [&](int i) { return (const float*)d_in[i]; };
  int c = 2;

  LevelP enc[2], dec[2];
  BlockP bridge[9];
  const float *patch_w, *patch_b, *patch_ln_g, *patch_ln_b, *head_w, *head_b;

  const bool insertion_order = (in_sizes[2] == 49152);  // patch_w first?
  if (insertion_order) {
    auto rdB = [&](BlockP& bp) {
      bp.ln_g = F(c++); bp.ln_b = F(c++); bp.in_w = F(c++); bp.conv_w = F(c++);
      bp.conv_b = F(c++); bp.xproj_w = F(c++); bp.dt_w = F(c++); bp.dt_b = F(c++);
      bp.A_log = F(c++); bp.Dp = F(c++); bp.out_w = F(c++);
    };
    auto rdL = [&](LevelP& L) {
      rdB(L.blk[0]); rdB(L.blk[1]);
      L.lin_w = F(c++); L.lin_b = F(c++); L.ln_g = F(c++); L.ln_b = F(c++);
    };
    patch_w = F(c++); patch_b = F(c++); patch_ln_g = F(c++); patch_ln_b = F(c++);
    rdL(enc[0]); rdL(enc[1]);
    for (int i = 0; i < 9; ++i) rdB(bridge[i]);
    rdL(dec[0]); rdL(dec[1]);
    head_w = F(c++); head_b = F(c++);
  } else {  // jax sorted-key pytree order
    auto rdB = [&](BlockP& bp) {
      bp.A_log = F(c++); bp.Dp = F(c++); bp.conv_b = F(c++); bp.conv_w = F(c++);
      bp.dt_b = F(c++); bp.dt_w = F(c++); bp.in_w = F(c++); bp.ln_b = F(c++);
      bp.ln_g = F(c++); bp.out_w = F(c++); bp.xproj_w = F(c++);
    };
    auto rdL = [&](LevelP& L) {
      rdB(L.blk[0]); rdB(L.blk[1]);
      L.lin_b = F(c++); L.lin_w = F(c++); L.ln_b = F(c++); L.ln_g = F(c++);
    };
    for (int i = 0; i < 9; ++i) rdB(bridge[i]);
    rdL(dec[0]); rdL(dec[1]);
    rdL(enc[0]); rdL(enc[1]);
    head_b = F(c++); head_w = F(c++);
    patch_b = F(c++); patch_ln_b = F(c++); patch_ln_g = F(c++); patch_w = F(c++);
  }

  // ---- workspace layout (256B aligned bump allocator) ----
  char* p = (char*)d_ws;
  auto bump = [&](size_t bytes) -> void* {
    void* r = (void*)p;
    p += (bytes + 255) & ~(size_t)255;
    return r;
  };
  _Float16* wt16   = (_Float16*)bump((size_t)1024 * 256 * 2);   // largest weight (f16, transposed)
  _Float16* h16    = (_Float16*)bump((size_t)TOK * DMODEL * 2); // also reused for pe16
  float*    xz32   = (float*)bump((size_t)TOK * 2 * DINNER * 4);
  float*    xm32   = (float*)bump((size_t)TOK * DINNER * 4);
  _Float16* xm16   = (_Float16*)bump((size_t)TOK * DINNER * 2);
  float*    xdbl32 = (float*)bump((size_t)TOK * XPROJ_N * 4);   // also reused for head out
  float*    dlt32  = (float*)bump((size_t)TOK * DINNER * 4);
  _Float16* yg16   = (_Float16*)bump((size_t)TOK * DINNER * 2);
  float*    seq    = (float*)bump((size_t)TOK * DMODEL * 4);
  float*    tmp32  = (float*)bump((size_t)TOK * DMODEL * 4);
  float*    resid0 = (float*)bump((size_t)TOK * DMODEL * 4);
  float*    resid1 = (float*)bump((size_t)TOK * DMODEL * 4);

  auto convT = [&](const float* W, int K, int N) {
    int n = K * N;
    convT_kernel<<<(n + 255) / 256, 256, 0, stream>>>(W, wt16, K, N);
  };
  auto gemm = [&](const _Float16* A, const float* bias, const float* Cadd,
                  float* C, int N, int K) {
    gemm_wmma<<<dim3(N / 16, TOK / 64), 128, 0, stream>>>(A, wt16, bias, Cadd, C, N, K);
  };

  auto runBlock = [&](const BlockP& bp) {
    ln_kernel<<<TOK, DMODEL, 0, stream>>>(seq, bp.ln_g, bp.ln_b, nullptr, h16, 0);
    convT(bp.in_w, DMODEL, 2 * DINNER);
    gemm(h16, nullptr, nullptr, xz32, 2 * DINNER, DMODEL);
    dwconv_kernel<<<(TOK * DINNER + 255) / 256, 256, 0, stream>>>(
        xz32, bp.conv_w, bp.conv_b, xm32, xm16);
    convT(bp.xproj_w, DINNER, XPROJ_N);
    gemm(xm16, nullptr, nullptr, xdbl32, XPROJ_N, DINNER);
    delta_kernel<<<(TOK * DINNER + 255) / 256, 256, 0, stream>>>(
        xdbl32, bp.dt_w, bp.dt_b, dlt32);
    scan_kernel<<<(BATCH * DINNER + 127) / 128, 128, 0, stream>>>(
        xm32, dlt32, xdbl32, bp.A_log, bp.Dp, xz32, yg16);
    convT(bp.out_w, DINNER, DMODEL);
    gemm(yg16, nullptr, seq, seq, DMODEL, DINNER);  // fused residual add
  };
  auto runLevel = [&](const LevelP& L) {
    runBlock(L.blk[0]);
    runBlock(L.blk[1]);
    cast16_kernel<<<(TOK * DMODEL + 255) / 256, 256, 0, stream>>>(seq, h16, TOK * DMODEL);
    convT(L.lin_w, DMODEL, DMODEL);
    gemm(h16, L.lin_b, nullptr, tmp32, DMODEL, DMODEL);
    ln_kernel<<<TOK, DMODEL, 0, stream>>>(tmp32, L.ln_g, L.ln_b, seq, nullptr, 1);
  };

  // ---- patch embedding ----
  patch_kernel<<<(TOK * 192 + 255) / 256, 256, 0, stream>>>(x_in, fx_in, h16);
  convT(patch_w, 192, DMODEL);
  gemm(h16, patch_b, nullptr, tmp32, DMODEL, 192);
  ln_kernel<<<TOK, DMODEL, 0, stream>>>(tmp32, patch_ln_g, patch_ln_b, seq, nullptr, 0);

  // ---- encoder ----
  runLevel(enc[0]);
  hipMemcpyAsync(resid0, seq, (size_t)TOK * DMODEL * 4, hipMemcpyDeviceToDevice, stream);
  runLevel(enc[1]);
  hipMemcpyAsync(resid1, seq, (size_t)TOK * DMODEL * 4, hipMemcpyDeviceToDevice, stream);

  // ---- bridge ----
  for (int i = 0; i < 9; ++i) runBlock(bridge[i]);

  // ---- decoder ----
  add_kernel<<<(TOK * DMODEL + 255) / 256, 256, 0, stream>>>(seq, resid1, TOK * DMODEL);
  runLevel(dec[0]);
  add_kernel<<<(TOK * DMODEL + 255) / 256, 256, 0, stream>>>(seq, resid0, TOK * DMODEL);
  runLevel(dec[1]);

  // ---- head ----
  cast16_kernel<<<(TOK * DMODEL + 255) / 256, 256, 0, stream>>>(seq, h16, TOK * DMODEL);
  convT(head_w, DMODEL, 16);
  gemm(h16, nullptr, nullptr, xdbl32, 16, DMODEL);
  head_kernel<<<(BATCH * 64 * 64 + 255) / 256, 256, 0, stream>>>(
      xdbl32, head_b, (float*)d_out);
}
